// SoftArgMax2d_76519137345901
// MI455X (gfx1250) — matrix operand order repaired and checked by reading, not compiled, not run on hardware
//
#include <hip/hip_runtime.h>
#include <hip/hip_bf16.h>

#define B_ 64
#define C_ 64
#define H_ 128
#define W_ 128

typedef __attribute__((ext_vector_type(16))) _Float16 v16h;
typedef __attribute__((ext_vector_type(8)))  _Float16 h8;
typedef __attribute__((ext_vector_type(4)))  _Float16 h4;
typedef __attribute__((ext_vector_type(8)))  float    v8f;

// Convert four float4s (16 floats) into a v16h fragment, scaling by m.
__device__ __forceinline__ v16h pack16(float4 a, float4 b, float4 c, float4 d, float m) {
    v16h r;
    r[0]  = (_Float16)(a.x * m); r[1]  = (_Float16)(a.y * m);
    r[2]  = (_Float16)(a.z * m); r[3]  = (_Float16)(a.w * m);
    r[4]  = (_Float16)(b.x * m); r[5]  = (_Float16)(b.y * m);
    r[6]  = (_Float16)(b.z * m); r[7]  = (_Float16)(b.w * m);
    r[8]  = (_Float16)(c.x * m); r[9]  = (_Float16)(c.y * m);
    r[10] = (_Float16)(c.z * m); r[11] = (_Float16)(c.w * m);
    r[12] = (_Float16)(d.x * m); r[13] = (_Float16)(d.y * m);
    r[14] = (_Float16)(d.z * m); r[15] = (_Float16)(d.w * m);
    return r;
}

// ---------------------------------------------------------------------------
// Kernel 1: one block per (b,c) image. Single HBM pass.
// Uses sum_h exp(x-m) == exp(-m) * sum_h exp(x): track max with v_max_f32,
// accumulate exp(x) branch-free (1 trans/elem), scale once at the end.
//   E[w] = sum_h exp(x[h,w]-m),  F[w] = sum_h exp(x[h,w]-m) * (h/127)
// Thread t owns columns 4g..4g+3 (g=t&31), rows seg*16..+15 (seg=t>>5);
// float4 loads -> 512B contiguous per wave issue.
// ---------------------------------------------------------------------------
__global__ __launch_bounds__(256) void sam2d_colsums(const float* __restrict__ x,
                                                     float* __restrict__ E,
                                                     float* __restrict__ F) {
    const int bc = blockIdx.x;                       // (b*C + c)
    const float* img = x + (size_t)bc * (H_ * W_);
    const int t   = threadIdx.x;
    const int g   = t & 31;                          // column group (4 cols)
    const int seg = t >> 5;                          // row segment (16 rows)

    float mw = -3.402823466e38f;
    float sE[4] = {0.f, 0.f, 0.f, 0.f};
    float sF[4] = {0.f, 0.f, 0.f, 0.f};

    const float inv127 = 1.0f / 127.0f;
#pragma unroll
    for (int r = 0; r < 16; ++r) {
        const int h = seg * 16 + r;
        const float4 v = *(const float4*)(img + (size_t)h * W_ + g * 4);
        const float wy = (float)h * inv127;
        const float vv[4] = { v.x, v.y, v.z, v.w };
#pragma unroll
        for (int j = 0; j < 4; ++j) {
            mw = fmaxf(mw, vv[j]);
            const float e = __expf(vv[j]);
            sE[j] += e;
            sF[j] += e * wy;
        }
    }

    // Block-wide max.
    __shared__ float red[256];
    red[t] = mw;
    __syncthreads();
    for (int s = 128; s > 0; s >>= 1) {
        if (t < s) red[t] = fmaxf(red[t], red[t + s]);
        __syncthreads();
    }
    const float scale = __expf(-red[0]);             // exp(-m)

    // Combine the 8 row segments per column.
    __shared__ float segE[8][W_];
    __shared__ float segF[8][W_];
#pragma unroll
    for (int j = 0; j < 4; ++j) {
        segE[seg][g * 4 + j] = sE[j] * scale;
        segF[seg][g * 4 + j] = sF[j] * scale;
    }
    __syncthreads();
    if (t < W_) {
        float e = 0.f, f = 0.f;
#pragma unroll
        for (int s2 = 0; s2 < 8; ++s2) { e += segE[s2][t]; f += segF[s2][t]; }
        E[(size_t)bc * W_ + t] = e;
        F[(size_t)bc * W_ + t] = f;
    }
}

// ---------------------------------------------------------------------------
// Kernel 2: one block per batch b, 128 threads = 4 waves (wave32).
//   s[w] = sum_c E[b,c,w];  qx[w] = (w/127)/s[w];  qy[w] = 1/s[w]
//   xx[c] = sum_w E[c,w]*qx[w],  xy[c] = sum_w F[c,w]*qy[w]
// V_WMMA_F32_16X16X32_F16 per wave over 16 channels; branch-free operand
// construction, all LDS traffic as ds_load_b128.
// ---------------------------------------------------------------------------
__global__ __launch_bounds__(128) void sam2d_wmma_contract(const float* __restrict__ E,
                                                           const float* __restrict__ F,
                                                           float* __restrict__ out) {
    const int b    = blockIdx.x;
    const int t    = threadIdx.x;
    const int lane = t & 31;
    const int wave = t >> 5;                       // channels wave*16 .. +15

    __shared__ float     lE[C_ * W_];              // 32 KB fp32 (feeds s and A_E)
    __shared__ _Float16  lF[C_ * W_];              // 16 KB f16  (feeds A_F)
    __shared__ float     q[2][W_];                 // qx, qy

    const float* Eb = E + (size_t)b * C_ * W_;
    const float* Fb = F + (size_t)b * C_ * W_;

    for (int i = t; i < (C_ * W_) / 4; i += 128) {
        const float4 ve = ((const float4*)Eb)[i];
        const float4 vf = ((const float4*)Fb)[i];
        ((float4*)lE)[i] = ve;
        h4 hv;
        hv[0] = (_Float16)vf.x; hv[1] = (_Float16)vf.y;
        hv[2] = (_Float16)vf.z; hv[3] = (_Float16)vf.w;
        ((h4*)lF)[i] = hv;
    }
    __syncthreads();

    {   // thread t owns column w = t (128 threads == W_)
        float s = 0.f;
        for (int c = 0; c < C_; ++c) s += lE[c * W_ + t];
        const float inv = 1.0f / s;
        q[0][t] = ((float)t * (1.0f / 127.0f)) * inv;   // qx = wx/s
        q[1][t] = inv;                                   // qy = 1/s
    }
    __syncthreads();

    const int   mrow  = lane & 15;
    const int   hi    = lane >> 4;                 // 0: lanes 0-15, 1: lanes 16-31
    const int   chan  = wave * 16 + mrow;
    const float bmask = (mrow == 0) ? 1.0f : 0.0f; // column-0-only B, branch-free

    v8f accx = {}; v8f accy = {};
#pragma unroll
    for (int kc = 0; kc < 4; ++kc) {
        const int kbase = kc * 32;
        // A 16x32 f16 layout per lane: two contiguous runs of 8,
        // K = hi*8 + {0..7} and hi*8 + 16 + {0..7}.
        const int abase = chan * W_ + kbase + hi * 8;
        const float4 e0 = *(const float4*)&lE[abase +  0];
        const float4 e1 = *(const float4*)&lE[abase +  4];
        const float4 e2 = *(const float4*)&lE[abase + 16];
        const float4 e3 = *(const float4*)&lE[abase + 20];
        const v16h aE = pack16(e0, e1, e2, e3, 1.0f);

        const h8 f0 = *(const h8*)&lF[abase];
        const h8 f1 = *(const h8*)&lF[abase + 16];
        const v16h aF = __builtin_shufflevector(f0, f1, 0, 1, 2, 3, 4, 5, 6, 7,
                                                8, 9, 10, 11, 12, 13, 14, 15);

        // B 32x16 f16 layout per lane: two contiguous runs of 8,
        // K = hi*8 + {0..7} and hi*8 + 16 + {0..7}; only column N==0 (mrow==0) is nonzero.
        const int qb = kbase + hi * 8;
        const float4 x0 = *(const float4*)&q[0][qb +  0];
        const float4 x1 = *(const float4*)&q[0][qb +  4];
        const float4 x2 = *(const float4*)&q[0][qb + 16];
        const float4 x3 = *(const float4*)&q[0][qb + 20];
        const v16h bx = pack16(x0, x1, x2, x3, bmask);

        const float4 y0 = *(const float4*)&q[1][qb +  0];
        const float4 y1 = *(const float4*)&q[1][qb +  4];
        const float4 y2 = *(const float4*)&q[1][qb + 16];
        const float4 y3 = *(const float4*)&q[1][qb + 20];
        const v16h by = pack16(y0, y1, y2, y3, bmask);

        accx = __builtin_amdgcn_wmma_f32_16x16x32_f16(false, aE, false, bx,
                                                      (short)0, accx, false, false);
        accy = __builtin_amdgcn_wmma_f32_16x16x32_f16(false, aF, false, by,
                                                      (short)0, accy, false, false);
    }

    // D column N=0 lives in lanes 0 (M=0..7 -> acc[0..7]) and 16 (M=8..15).
    if (mrow == 0) {
        const int chbase = wave * 16 + hi * 8;
#pragma unroll
        for (int r = 0; r < 8; ++r) {
            const int ch = chbase + r;
            out[((size_t)b * C_ + ch) * 2 + 0] = accx[r];
            out[((size_t)b * C_ + ch) * 2 + 1] = accy[r];
        }
    }
}

extern "C" void kernel_launch(void* const* d_in, const int* in_sizes, int n_in,
                              void* d_out, int out_size, void* d_ws, size_t ws_size,
                              hipStream_t stream) {
    (void)in_sizes; (void)n_in; (void)out_size; (void)ws_size;
    const float* x = (const float*)d_in[0];
    float* out = (float*)d_out;
    float* E = (float*)d_ws;                          // [B,C,W] fp32 (2 MB)
    float* F = E + (size_t)B_ * C_ * W_;              // [B,C,W] fp32 (2 MB)

    sam2d_colsums<<<B_ * C_, 256, 0, stream>>>(x, E, F);
    sam2d_wmma_contract<<<B_, 128, 0, stream>>>(E, F, out);
}
